// LSTM_7627861918399
// MI455X (gfx1250) — matrix operand order repaired and checked by reading, not compiled, not run on hardware
//
#include <hip/hip_runtime.h>
#include <math.h>

typedef __bf16 bf16;
typedef __attribute__((ext_vector_type(8)))  __bf16 v8bf;
typedef __attribute__((ext_vector_type(16))) __bf16 v16bf;
typedef __attribute__((ext_vector_type(8)))  float  v8f;

#define HID     1024
#define BATCH   32
#define TSTEPS  256
#define GCOLS   4096
#define NTILES  256   // GCOLS / 16

static __device__ __forceinline__ v16bf cat16(v8bf lo, v8bf hi) {
    return __builtin_shufflevector(lo, hi, 0,1,2,3,4,5,6,7,8,9,10,11,12,13,14,15);
}

// A tile: lane holds rows of 16-bit A striping; two 16B chunks at +0 / +16 elems
static __device__ __forceinline__ v16bf load_a(const bf16* p) {
    v8bf lo = *(const v8bf*)p;
    v8bf hi = *(const v8bf*)(p + 16);
    return cat16(lo, hi);
}
// B tile from packed layout: 16 contiguous bf16 per lane
static __device__ __forceinline__ v16bf load_b(const bf16* p) {
    v8bf lo = *(const v8bf*)p;
    v8bf hi = *(const v8bf*)(p + 8);
    return cat16(lo, hi);
}

static __device__ __forceinline__ v8f wmma_bf16(v16bf a, v16bf b, v8f c) {
    return __builtin_amdgcn_wmma_f32_16x16x32_bf16(false, a, false, b, (short)0, c, false, false);
}

// branchless fast sigmoid / tanh (single v_exp_f32 each, no EXEC divergence)
static __device__ __forceinline__ float sigm(float x) {
    return __fdividef(1.0f, 1.0f + __expf(-x));
}
static __device__ __forceinline__ float fast_tanh(float x) {
    float e = __expf(-2.0f * fabsf(x));
    float r = __fdividef(1.0f - e, 1.0f + e);
    return copysignf(r, x);
}

// ---------------- grid-wide barrier (persistent kernel) ----------------
struct GBar { unsigned cnt; unsigned gen; };

static __device__ __forceinline__ void grid_barrier(GBar* b, unsigned nblocks) {
    __syncthreads();
    if (threadIdx.x == 0) {
        __threadfence();
        unsigned gen = __hip_atomic_load(&b->gen, __ATOMIC_RELAXED, __HIP_MEMORY_SCOPE_AGENT);
        unsigned prev = __hip_atomic_fetch_add(&b->cnt, 1u, __ATOMIC_ACQ_REL, __HIP_MEMORY_SCOPE_AGENT);
        if (prev == nblocks - 1) {
            __hip_atomic_store(&b->cnt, 0u, __ATOMIC_RELAXED, __HIP_MEMORY_SCOPE_AGENT);
            __hip_atomic_fetch_add(&b->gen, 1u, __ATOMIC_RELEASE, __HIP_MEMORY_SCOPE_AGENT);
        } else {
            while (__hip_atomic_load(&b->gen, __ATOMIC_ACQUIRE, __HIP_MEMORY_SCOPE_AGENT) == gen) {
                __builtin_amdgcn_s_sleep(2);
            }
        }
    }
    __syncthreads();
}

__global__ void init_bar_kernel(GBar* b) { b->cnt = 0u; b->gen = 0u; }

// ---------------- weight pack: fp32 [K x 4096] -> bf16 WMMA-B layout ----
// Output elem o = ((kt*256 + nt)*32 + lane)*16 + i
// lane holds column (lane&15), K = kt*32 + (lane>>4)*8 + (i&7) + (i>>3)*16   (16 bf16/lane)
// Column permutation: original co = g*1024 + nh  ->  p = (nh/16)*64 + g*16 + (nh%16)
__global__ void pack_w_kernel(const float* __restrict__ W, int krows, bf16* __restrict__ out) {
    int total = krows * GCOLS;
    for (int o = blockIdx.x * blockDim.x + threadIdx.x; o < total; o += gridDim.x * blockDim.x) {
        int i    = o & 15;
        int lane = (o >> 4) & 31;
        int nt   = (o >> 9) & 255;
        int kt   = o >> 17;
        int k    = kt * 32 + (lane >> 4) * 8 + (i & 7) + (i >> 3) * 16;
        int p    = nt * 16 + (lane & 15);
        int blk  = p >> 6, r = p & 63;
        int g    = r >> 4, nl = r & 15;
        int co   = g * HID + blk * 16 + nl;
        out[o] = (bf16)W[(long)k * GCOLS + co];
    }
}

// bias: permute columns, fold forget bias (gate f == 2) in.
__global__ void pack_bias_kernel(const float* __restrict__ b, float* __restrict__ out) {
    int p = blockIdx.x * blockDim.x + threadIdx.x;
    if (p < GCOLS) {
        int blk = p >> 6, r = p & 63;
        int g = r >> 4, nl = r & 15;
        int co = g * HID + blk * 16 + nl;
        out[p] = b[co] + (g == 2 ? 1.0f : 0.0f);
    }
}

__global__ void f2bf_kernel(const float* __restrict__ in, bf16* __restrict__ out, int n) {
    for (int i = blockIdx.x * blockDim.x + threadIdx.x; i < n; i += gridDim.x * blockDim.x)
        out[i] = (bf16)in[i];
}

// ---------------- big precompute GEMM:  C[8192 x 4096] = A @ BP + bias ----
// block = 256 thr = 8 waves; each wave computes a 32x32 register tile (2x2
// WMMA tiles) for 2x A/B reuse; block covers 64 rows x 128 cols.
// Software-pipelined: loads for kt+1 issued before the WMMAs of kt.
template <int K>
__global__ void __launch_bounds__(256) gemm_zx_kernel(
        const bf16* __restrict__ A, int lda,
        const bf16* __restrict__ BP, const float* __restrict__ bias,
        float* __restrict__ C) {
    int mblk = blockIdx.x;                 // 128 blocks * 64 rows
    int nblk = blockIdx.y;                 // 32 blocks * 128 cols
    int wave = threadIdx.x >> 5, lane = threadIdx.x & 31;
    int wm = wave >> 2;                    // 0..1 : 32-row slab
    int wn = wave & 3;                     // 0..3 : 32-col slab
    int mbase  = mblk * 64 + wm * 32;
    int ntile0 = (nblk * 128 + wn * 32) >> 4;   // first of two 16-col tiles
    int row0   = mbase + (lane & 15);
    int khalf  = (lane >> 4) * 8;

    const long kstride = (long)NTILES << 9;      // packed elems per k-tile
    const bf16* pa0 = A + (long)row0 * lda + khalf;
    const bf16* pa1 = pa0 + (long)16 * lda;
    const bf16* pb0 = BP + ((long)ntile0 << 9) + (lane << 4);
    const bf16* pb1 = pb0 + 512;

    v8f acc[4] = {};
    const int KT = K >> 5;

    v16bf ca0 = load_a(pa0), ca1 = load_a(pa1);
    v16bf cb0 = load_b(pb0), cb1 = load_b(pb1);
#pragma unroll 4
    for (int kt = 0; kt < KT; ++kt) {
        int ktn = (kt + 1 < KT) ? kt + 1 : kt;   // last iter reloads (discarded)
        v16bf na0 = load_a(pa0 + ktn * 32);
        v16bf na1 = load_a(pa1 + ktn * 32);
        v16bf nb0 = load_b(pb0 + (long)ktn * kstride);
        v16bf nb1 = load_b(pb1 + (long)ktn * kstride);
        acc[0] = wmma_bf16(ca0, cb0, acc[0]);
        acc[1] = wmma_bf16(ca0, cb1, acc[1]);
        acc[2] = wmma_bf16(ca1, cb0, acc[2]);
        acc[3] = wmma_bf16(ca1, cb1, acc[3]);
        ca0 = na0; ca1 = na1; cb0 = nb0; cb1 = nb1;
    }

#pragma unroll
    for (int i = 0; i < 2; ++i) {
#pragma unroll
        for (int j = 0; j < 2; ++j) {
            int col = (ntile0 + j) * 16 + (lane & 15);
            float bs = bias[col];
            int rbase = mbase + i * 16 + (lane >> 4) * 8;
            v8f a = acc[i * 2 + j];
#pragma unroll
            for (int r = 0; r < 8; ++r)
                C[(long)(rbase + r) * GCOLS + col] = a[r] + bs;
        }
    }
}

// ---------------- persistent sequential LSTM layer ----------------
// 64 blocks x 256 threads. Block blk owns h columns [blk*16, blk*16+16) = all 4
// gates at permuted cols [blk*64, blk*64+64). Recurrent weight slice (128 KB)
// lives in LDS for the whole sequence.
__global__ void __launch_bounds__(256, 1) lstm_seq_kernel(
        const float* __restrict__ zx,     // [8192 x 4096] permuted, row = b*T + t
        const bf16* __restrict__ WhP,     // packed [1024 x 4096]
        bf16* __restrict__ hcur,          // double buffer 2 x [32 x 1024]
        bf16* __restrict__ h_all,         // [8192 x 1024] bf16 (nullable)
        float* __restrict__ out_f32,      // [8192 x 1024] f32  (nullable)
        GBar* bar) {
    __shared__ bf16  sW[32 * 4 * 512];    // 128 KB: 32 k-tiles x 4 gate-tiles
    __shared__ float sG[4 * 512];         // 8 KB: gate pre-activations

    int tid = threadIdx.x, blk = blockIdx.x;
    int wave = tid >> 5, lane = tid & 31;
    int mt = wave >> 2, g = wave & 3;

    // stage recurrent weights into LDS (8192 chunks of 16 B)
    for (int ci = tid; ci < 8192; ci += 256) {
        int c2 = ci & 1, ln = (ci >> 1) & 31, lt = ci >> 6;
        int kt = lt >> 2, gg = lt & 3;
        const bf16* src = WhP + ((long)(kt * NTILES + blk * 4 + gg) << 9) + (ln << 4) + (c2 << 3);
        *(v8bf*)(sW + (lt << 9) + (ln << 4) + (c2 << 3)) = *(const v8bf*)src;
    }
    // zero h (buffer 0) for this block's column slice
    for (int e = tid; e < 512; e += 256) {
        int b = e >> 4, nl = e & 15;
        hcur[b * HID + blk * 16 + nl] = (bf16)0.0f;
    }
    float creg[2] = {0.0f, 0.0f};
    grid_barrier(bar, gridDim.x);

    int col   = blk * 64 + g * 16 + (lane & 15);
    int rb    = mt * 16 + (lane >> 4) * 8;
    int arow  = mt * 16 + (lane & 15);
    int khalf = (lane >> 4) * 8;
    const bf16* sWg = sW + ((long)g << 9) + (lane << 4);

    for (int t = 0; t < TSTEPS; ++t) {
        const bf16* hprev = hcur + (t & 1) * (BATCH * HID);
        bf16* hnext = hcur + ((t + 1) & 1) * (BATCH * HID);
        const bf16* ap = hprev + arow * HID + khalf;

        v8f acc;
#pragma unroll
        for (int r = 0; r < 8; ++r)
            acc[r] = zx[(long)((rb + r) * TSTEPS + t) * GCOLS + col];

        // prefetch next step's zx slice while this step computes
        if (t + 1 < TSTEPS) {
#pragma unroll
            for (int r = 0; r < 8; ++r)
                __builtin_prefetch(&zx[(long)((rb + r) * TSTEPS + t + 1) * GCOLS + col], 0, 0);
        }

#pragma unroll 8
        for (int kt = 0; kt < 32; ++kt) {
            v16bf a = load_a(ap + (kt << 5));
            v16bf b = load_b(sWg + ((long)kt << 11));   // (kt*4)<<9
            acc = wmma_bf16(a, b, acc);
        }
#pragma unroll
        for (int r = 0; r < 8; ++r)
            sG[g * 512 + (rb + r) * 16 + (lane & 15)] = acc[r];
        __syncthreads();

#pragma unroll
        for (int u = 0; u < 2; ++u) {
            int e = tid + u * 256;
            int b = e >> 4, nl = e & 15;
            float gi = sG[          b * 16 + nl];
            float gj = sG[ 512 +    b * 16 + nl];
            float gf = sG[1024 +    b * 16 + nl];
            float go = sG[1536 +    b * 16 + nl];
            float cn = sigm(gf) * creg[u] + sigm(gi) * fast_tanh(gj);
            float h  = sigm(go) * fast_tanh(cn);
            creg[u] = cn;
            int nh = blk * 16 + nl;
            hnext[b * HID + nh] = (bf16)h;
            long orow = (long)(b * TSTEPS + t) * HID + nh;
            if (h_all)   h_all[orow] = (bf16)h;
            if (out_f32) out_f32[orow] = h;
        }
        grid_barrier(bar, gridDim.x);  // publish h for step t+1
    }
}

// ---------------- host-side launcher ----------------
extern "C" void kernel_launch(void* const* d_in, const int* in_sizes, int n_in,
                              void* d_out, int out_size, void* d_ws, size_t ws_size,
                              hipStream_t stream) {
    const float* x  = (const float*)d_in[0];   // [32,256,512]
    const float* W1 = (const float*)d_in[1];   // [1536,4096]
    const float* b1 = (const float*)d_in[2];   // [4096]
    const float* W2 = (const float*)d_in[3];   // [2048,4096]
    const float* b2 = (const float*)d_in[4];   // [4096]
    float* out = (float*)d_out;                // [32,256,1024]

    char* ws = (char*)d_ws;
    size_t off = 0;
    auto alloc = [&](size_t bytes) -> void* {
        void* p = ws + off;
        off = (off + bytes + 255) & ~(size_t)255;
        return p;
    };
    bf16*  W1xP = (bf16*)alloc((size_t)512  * 4096 * 2);
    bf16*  W1hP = (bf16*)alloc((size_t)1024 * 4096 * 2);
    bf16*  W2xP = (bf16*)alloc((size_t)1024 * 4096 * 2);
    bf16*  W2hP = (bf16*)alloc((size_t)1024 * 4096 * 2);
    float* b1p  = (float*)alloc(4096 * 4);
    float* b2p  = (float*)alloc(4096 * 4);
    bf16*  xbf  = (bf16*)alloc((size_t)8192 * 512 * 2);
    bf16*  h1a  = (bf16*)alloc((size_t)8192 * 1024 * 2);
    float* zx   = (float*)alloc((size_t)8192 * 4096 * 4);  // reused by both layers
    bf16*  hcur = (bf16*)alloc((size_t)2 * 32 * 1024 * 2);
    GBar*  bar  = (GBar*)alloc(256);

    // pack weights (bf16, WMMA-B striping, gate-interleaved column permutation)
    pack_w_kernel<<<2048, 256, 0, stream>>>(W1,               512,  W1xP);
    pack_w_kernel<<<4096, 256, 0, stream>>>(W1 + 512  * 4096, 1024, W1hP);
    pack_w_kernel<<<4096, 256, 0, stream>>>(W2,               1024, W2xP);
    pack_w_kernel<<<4096, 256, 0, stream>>>(W2 + 1024 * 4096, 1024, W2hP);
    pack_bias_kernel<<<16, 256, 0, stream>>>(b1, b1p);
    pack_bias_kernel<<<16, 256, 0, stream>>>(b2, b2p);
    f2bf_kernel<<<4096, 256, 0, stream>>>(x, xbf, 8192 * 512);
    init_bar_kernel<<<1, 1, 0, stream>>>(bar);

    // layer 1: parallel input GEMM, then persistent recurrence
    gemm_zx_kernel<512><<<dim3(128, 32), 256, 0, stream>>>(xbf, 512, W1xP, b1p, zx);
    lstm_seq_kernel<<<64, 256, 0, stream>>>(zx, W1hP, hcur, h1a, (float*)nullptr, bar);

    // layer 2: input GEMM over h1, then recurrence writing final f32 output
    gemm_zx_kernel<1024><<<dim3(128, 32), 256, 0, stream>>>(h1a, 1024, W2xP, b2p, zx);
    lstm_seq_kernel<<<64, 256, 0, stream>>>(zx, W2hP, hcur, (bf16*)nullptr, out, bar);
}
